// CrossDeformAttn_45784351375777
// MI455X (gfx1250) — matrix operand order repaired and checked by reading, not compiled, not run on hardware
//
// CrossDeformAttn for MI455X (gfx1250, wave32, WMMA + TDM).
// Pipeline: LN -> q GEMM (WMMA+TDM, double-buffered) -> collapsed 1D branch
//           -> 2D offsets/bilinear -> per-patch 16x16 WMMA attention
//           -> write GEMM -> proj GEMM.  B operands are stored N-major (BT)
//           so both A and B WMMA fragments are two ds_load_b128 each.
#include <hip/hip_runtime.h>

typedef _Float16 half_t;
typedef __attribute__((ext_vector_type(16))) _Float16 v16h;
typedef __attribute__((ext_vector_type(8)))  float    v8f;
typedef __attribute__((ext_vector_type(4)))  unsigned int u32x4;
typedef __attribute__((ext_vector_type(8)))  int          i32x8;
typedef __attribute__((ext_vector_type(4)))  int          i32x4;

#ifndef USE_TDM
#if defined(__AMDGCN__) && defined(__has_builtin)
#if __has_builtin(__builtin_amdgcn_tensor_load_to_lds) && __has_builtin(__builtin_amdgcn_s_wait_tensorcnt)
#define USE_TDM 1
#else
#define USE_TDM 0
#endif
#else
#define USE_TDM 0
#endif
#endif

#define GBM 128
#define GBN 64
#define GBK 32

#if USE_TDM
// Build a D# (groups 0/1 per CDNA5 ISA) for a 2-D f16
// tile: tile_w contiguous elements per row, tile_h rows, row pitch row_stride
// elements; destination is contiguous LDS at lds_off.  6-arg builtin:
//   (u32x4 g0, i32x8 g1, i32x4 g2, i32x4 g3, i32x8 spare, i32 cpol)
__device__ __forceinline__ void tdm_load_2d(const void* gptr, unsigned lds_off,
                                            unsigned tile_w, unsigned tile_h,
                                            unsigned row_stride) {
  unsigned long long ga = (unsigned long long)(size_t)gptr;
  u32x4 g0;
  g0[0] = 1u;                                        // count=1, load, no gather
  g0[1] = lds_off;                                   // lds_addr (bytes)
  g0[2] = (unsigned)(ga & 0xffffffffull);            // global_addr[31:0]
  g0[3] = (unsigned)((ga >> 32) & 0x1ffffffull) | (2u << 30); // addr[56:32] | type=2
  unsigned td0 = row_stride;                         // tensor_dim0 (in-bounds use)
  unsigned td1 = 0x7fffffffu;                        // tensor_dim1 (in-bounds use)
  unsigned long long st0 = (unsigned long long)row_stride; // tensor_dim0_stride
  i32x8 g1;
  g1[0] = (int)(1u << 16);                           // data_size=1 -> 2 bytes
  g1[1] = (int)((td0 & 0xffffu) << 16);
  g1[2] = (int)(((td0 >> 16) & 0xffffu) | ((td1 & 0xffffu) << 16));
  g1[3] = (int)(((td1 >> 16) & 0xffffu) | (tile_w << 16));   // tile_dim0
  g1[4] = (int)(tile_h & 0xffffu);                   // tile_dim1, tile_dim2=0
  g1[5] = (int)(st0 & 0xffffffffull);
  g1[6] = (int)((st0 >> 32) & 0xffffull);
  g1[7] = 0;
  i32x4 z4; z4[0] = 0; z4[1] = 0; z4[2] = 0; z4[3] = 0;
  i32x8 z8; z8[0] = 0; z8[1] = 0; z8[2] = 0; z8[3] = 0;
  z8[4] = 0; z8[5] = 0; z8[6] = 0; z8[7] = 0;
  __builtin_amdgcn_tensor_load_to_lds(g0, g1, z4, z4, z8, 0);
}
#endif

// ---------------- generic f16 WMMA GEMM:  C = A(MxK) * BT(NxK)^T (+bias) ----
// B is supplied TRANSPOSED (N-major) so B fragments are contiguous in LDS.
// Requires M%128==0, N%64==0, K%32==0.  256 threads = 8 waves; wave w computes
// rows [w*16, w*16+16) x all 64 cols (4 accumulators, A fragment reused 4x).
__global__ __launch_bounds__(256) void gemm_wmma_kernel(
    const half_t* __restrict__ A, int lda, long long sA,
    const half_t* __restrict__ BT, int ldb, long long sB,
    half_t* __restrict__ Ch, float* __restrict__ Cf, int ldc, long long sC,
    const float* __restrict__ colBias, const float* __restrict__ rowBias,
    int M, int N, int K) {
  __shared__ __align__(16) half_t As[2][GBM][GBK];
  __shared__ __align__(16) half_t Bs[2][GBN][GBK];
  const int bz = blockIdx.z;
  A += (long long)bz * sA;
  BT += (long long)bz * sB;
  const int m0 = blockIdx.y * GBM;
  const int n0 = blockIdx.x * GBN;
  const int tid = threadIdx.x;
  const int lane = tid & 31;
  const int wid = tid >> 5;              // 0..7 : 16-row strip
  const int r15 = lane & 15;
  const int alo = (lane < 16) ? 0 : 8;   // A k-slot base per ISA layout
  const int bk0 = (lane < 16) ? 0 : 8;   // B k base per ISA layout
  v8f zero = {};
  v8f acc[4] = {zero, zero, zero, zero};
  const int niter = K / GBK;
#if USE_TDM
  if (wid == 0) {                        // prologue: tile 0 into buffer 0
    tdm_load_2d(A + (long long)m0 * lda, (unsigned)(size_t)&As[0][0][0],
                GBK, GBM, (unsigned)lda);
    tdm_load_2d(BT + (long long)n0 * ldb, (unsigned)(size_t)&Bs[0][0][0],
                GBK, GBN, (unsigned)ldb);
    __builtin_amdgcn_s_wait_tensorcnt(0);
  }
  __syncthreads();
#endif
  for (int it = 0; it < niter; ++it) {
#if USE_TDM
    const int cur = it & 1;
    if (wid == 0 && it + 1 < niter) {    // overlap next-tile DMA with WMMAs
      int kb = (it + 1) * GBK;
      tdm_load_2d(A + (long long)m0 * lda + kb,
                  (unsigned)(size_t)&As[1 - cur][0][0], GBK, GBM, (unsigned)lda);
      tdm_load_2d(BT + (long long)n0 * ldb + kb,
                  (unsigned)(size_t)&Bs[1 - cur][0][0], GBK, GBN, (unsigned)ldb);
    }
#else
    const int cur = 0;
    const int kb = it * GBK;
    __syncthreads();
    {
      int r = tid >> 1, c = (tid & 1) * 16;     // A: 128 rows x 32 halves
      *(uint4*)&As[0][r][c] = *(const uint4*)(A + (long long)(m0 + r) * lda + kb + c);
      *(uint4*)&As[0][r][c + 8] = *(const uint4*)(A + (long long)(m0 + r) * lda + kb + c + 8);
      int r2 = tid >> 2, c2 = (tid & 3) * 8;    // BT: 64 rows x 32 halves
      *(uint4*)&Bs[0][r2][c2] = *(const uint4*)(BT + (long long)(n0 + r2) * ldb + kb + c2);
    }
    __syncthreads();
#endif
    union { uint4 u[2]; v16h v; } af;
    af.u[0] = *(const uint4*)&As[cur][wid * 16 + r15][alo];
    af.u[1] = *(const uint4*)&As[cur][wid * 16 + r15][alo + 16];
#pragma unroll
    for (int t = 0; t < 4; ++t) {
      union { uint4 u[2]; v16h v; } bf;
      bf.u[0] = *(const uint4*)&Bs[cur][t * 16 + r15][bk0];
      bf.u[1] = *(const uint4*)&Bs[cur][t * 16 + r15][bk0 + 16];
      acc[t] = __builtin_amdgcn_wmma_f32_16x16x32_f16(false, af.v, false, bf.v,
                                                      (short)0, acc[t], false, false);
    }
#if USE_TDM
    __syncthreads();                     // all waves done with buffer `cur`
    if (wid == 0 && it + 1 < niter) __builtin_amdgcn_s_wait_tensorcnt(0);
    __syncthreads();                     // publish buffer 1-cur
#endif
  }
  const int crow = wid * 16 + ((lane < 16) ? 0 : 8);
#pragma unroll
  for (int r = 0; r < 8; ++r) {
    long long m = m0 + crow + r;
#pragma unroll
    for (int t = 0; t < 4; ++t) {
      long long n = n0 + t * 16 + r15;
      float v = acc[t][r];
      if (colBias) v += colBias[n];
      if (rowBias) v += rowBias[m];
      long long off = (long long)bz * sC + m * ldc + n;
      if (Cf) Cf[off] = v;
      if (Ch) Ch[off] = (half_t)v;
    }
  }
}

// ---------------- small prep kernels ----------------------------------------
// transpose + convert: dst[n*K + k] = (f16)src[k*N + n]   (src is K x N)
__global__ void cvtT_f16_kernel(const float* __restrict__ src, half_t* __restrict__ dst,
                                int K, int N) {
  int i = blockIdx.x * blockDim.x + threadIdx.x;
  if (i >= N * K) return;
  int k = i % K, n = i / K;
  dst[i] = (half_t)src[(long long)k * N + n];
}

// write_w (2032,1024) -> transposed + K-padded (1024 x 2048) f16 (A operand)
__global__ void wwT_kernel(const float* __restrict__ ww, half_t* __restrict__ wwT) {
  int i = blockIdx.x * blockDim.x + threadIdx.x;
  if (i >= 1024 * 2048) return;
  int p = i & 2047, s = i >> 11;
  wwT[i] = (half_t)((p < 2032) ? ww[p * 1024 + s] : 0.0f);
}

// fold (off1_w @ off2_w) -> Wf[3][128], Wf[384] = off1_b . off2_w
__global__ void foldw_kernel(const float* __restrict__ o1w, const float* __restrict__ o1b,
                             const float* __restrict__ o2w, float* __restrict__ Wf) {
  int i = blockIdx.x * blockDim.x + threadIdx.x;
  if (i < 384) {
    float s = 0.f;
    for (int o = 0; o < 128; ++o) s += o1w[i * 128 + o] * o2w[o];
    Wf[i] = s;
  } else if (i == 384) {
    float s = 0.f;
    for (int o = 0; o < 128; ++o) s += o1b[o] * o2w[o];
    Wf[384] = s;
  }
}

// zero the K-padding columns (p=2032..2047) of yT (B operand of write GEMM)
__global__ void ypad_kernel(half_t* __restrict__ yT) {
  int i = blockIdx.x * blockDim.x + threadIdx.x;
  if (i >= 16 * 512 * 16) return;
  int pp = i & 15, d = (i >> 4) & 511, b = i >> 13;
  yT[((long long)b * 512 + d) * 2048 + 2032 + pp] = (half_t)0.0f;
}

// ---------------- LayerNorm (one 256-thread block per row) ------------------
__global__ __launch_bounds__(256) void ln_kernel(const float* __restrict__ x,
                                                 const float* __restrict__ w,
                                                 const float* __restrict__ bb,
                                                 float* __restrict__ xn,
                                                 half_t* __restrict__ xnh) {
  long long row = blockIdx.x;
  const float* xr = x + row * 512;
  int t = threadIdx.x, lane = t & 31, wid = t >> 5;
  float v0 = xr[t], v1 = xr[t + 256];
  __shared__ float sred[8];
  float s = v0 + v1;
#pragma unroll
  for (int m = 1; m < 32; m <<= 1) s += __shfl_xor(s, m, 32);
  if (lane == 0) sred[wid] = s;
  __syncthreads();
  float tot = 0.f;
#pragma unroll
  for (int i = 0; i < 8; ++i) tot += sred[i];
  float mu = tot * (1.0f / 512.0f);
  __syncthreads();
  float d0 = v0 - mu, d1 = v1 - mu;
  float q = d0 * d0 + d1 * d1;
#pragma unroll
  for (int m = 1; m < 32; m <<= 1) q += __shfl_xor(q, m, 32);
  if (lane == 0) sred[wid] = q;
  __syncthreads();
  float qt = 0.f;
#pragma unroll
  for (int i = 0; i < 8; ++i) qt += sred[i];
  float rstd = rsqrtf(qt * (1.0f / 512.0f) + 1e-5f);
  float o0 = d0 * rstd * w[t] + bb[t];
  float o1 = d1 * rstd * w[t + 256] + bb[t + 256];
  xn[row * 512 + t] = o0;       xn[row * 512 + t + 256] = o1;
  xnh[row * 512 + t] = (half_t)o0; xnh[row * 512 + t + 256] = (half_t)o1;
}

// ---------------- 1D branch: folded offsets -> tent weight fx ---------------
__global__ void fx_kernel(const float* __restrict__ q, const float* __restrict__ Wf,
                          float* __restrict__ fx) {
  int idx = blockIdx.x * blockDim.x + threadIdx.x;   // B*G*L = 65536
  if (idx >= 16 * 4 * 1024) return;
  int l = idx & 1023, g = (idx >> 10) & 3, b = idx >> 12;
  float s = Wf[384];
  for (int tt = 0; tt < 3; ++tt) {
    int ll = l + tt - 1;
    if (ll < 0 || ll >= 1024) continue;
    const float* qr = q + ((long long)(b * 1024 + ll)) * 512 + g * 128;
    const float* wr = Wf + tt * 128;
#pragma unroll 4
    for (int i = 0; i < 128; ++i) s += qr[i] * wr[i];
  }
  float off = tanhf(s) * 3.0f;
  float vs = 2.0f * ((float)l + off) / 1023.0f - 1.0f;
  float ix = ((vs + 1.0f) * 1.0f - 1.0f) * 0.5f;     // W=1 grid_sample x coord
  float x0 = floorf(ix);
  float wx1 = ix - x0;
  float f = 0.f;
  if (x0 == 0.f) f += (1.0f - wx1);                  // corner x0 valid
  if (x0 + 1.0f == 0.f) f += wx1;                    // corner x0+1 valid
  fx[idx] = f;
}

// Kg/Vg: mid-row outer-product factors contracted with wk/wv
__global__ void kgvg_kernel(const float* __restrict__ xn, const float* __restrict__ wk,
                            const float* __restrict__ wv, float* __restrict__ Kg,
                            float* __restrict__ Vg) {
  int idx = blockIdx.x * blockDim.x + threadIdx.x;   // B*G*D = 32768
  if (idx >= 16 * 4 * 512) return;
  int e = idx & 511, g = (idx >> 9) & 3, b = idx >> 11;
  const float* r0 = xn + ((long long)(b * 1024 + 511)) * 512 + g * 128;
  const float* r1 = r0 + 512;
  float ak = 0.f, av = 0.f;
  for (int c = 0; c < 128; ++c) {
    float m = 0.5f * (r0[c] + r1[c]);
    ak += m * wk[(g * 128 + c) * 512 + e];
    av += m * wv[(g * 128 + c) * 512 + e];
  }
  Kg[idx] = ak; Vg[idx] = av;
}

// qf[b,g,d] = sum_l q*fx ; qs[b,d] = sum_l q
__global__ void qred_kernel(const float* __restrict__ q, const float* __restrict__ fx,
                            float* __restrict__ qf, float* __restrict__ qs) {
  int idx = blockIdx.x * blockDim.x + threadIdx.x;   // B*D = 8192
  if (idx >= 16 * 512) return;
  int d = idx & 511, b = idx >> 9;
  const float* fb = fx + b * 4096;
  float s0 = 0, s1 = 0, s2 = 0, s3 = 0, sp = 0;
  for (int l = 0; l < 1024; ++l) {
    float v = q[((long long)(b * 1024 + l)) * 512 + d];
    sp += v;
    s0 += v * fb[l]; s1 += v * fb[1024 + l]; s2 += v * fb[2048 + l]; s3 += v * fb[3072 + l];
  }
  qf[(b * 4 + 0) * 512 + d] = s0; qf[(b * 4 + 1) * 512 + d] = s1;
  qf[(b * 4 + 2) * 512 + d] = s2; qf[(b * 4 + 3) * 512 + d] = s3;
  qs[idx] = sp;
}

// 64x64 channel attention per (b,h); outputs attn@Vg and attn@bv
__global__ __launch_bounds__(64) void attn1d_kernel(
    const float* __restrict__ qf, const float* __restrict__ qs,
    const float* __restrict__ Kg, const float* __restrict__ bk,
    const float* __restrict__ Vg, const float* __restrict__ bv,
    float* __restrict__ AV, float* __restrict__ Ab) {
  int bh = blockIdx.x;            // 0..127
  int b = bh >> 3, h = bh & 7;
  int i = threadIdx.x;            // 0..63
  float qfi[4];
#pragma unroll
  for (int g = 0; g < 4; ++g) qfi[g] = qf[(b * 4 + g) * 512 + h * 64 + i];
  float qsi = qs[b * 512 + h * 64 + i];
  float sc[64];
  float mx = -1e30f;
#pragma unroll
  for (int j = 0; j < 64; ++j) {
    float s = qsi * bk[h * 64 + j];
#pragma unroll
    for (int g = 0; g < 4; ++g) s += qfi[g] * Kg[(b * 4 + g) * 512 + h * 64 + j];
    s *= 0.044194173824159216f;   // 512^-0.5
    sc[j] = s; mx = fmaxf(mx, s);
  }
  float sum = 0.f;
#pragma unroll
  for (int j = 0; j < 64; ++j) { sc[j] = __expf(sc[j] - mx); sum += sc[j]; }
  float inv = 1.0f / sum;
#pragma unroll
  for (int g = 0; g < 4; ++g) {
    float a = 0.f;
#pragma unroll
    for (int j = 0; j < 64; ++j) a += sc[j] * Vg[(b * 4 + g) * 512 + h * 64 + j];
    AV[(b * 4 + g) * 512 + h * 64 + i] = a * inv;
  }
  float ab = 0.f;
#pragma unroll
  for (int j = 0; j < 64; ++j) ab += sc[j] * bv[h * 64 + j];
  Ab[b * 512 + h * 64 + i] = ab * inv;
}

__global__ void wsum_kernel(const float* __restrict__ wo, float* __restrict__ wsum) {
  int e = blockIdx.x * blockDim.x + threadIdx.x;
  if (e >= 512) return;
  float s = 0.f;
  for (int d = 0; d < 512; ++d) s += wo[d * 512 + e];
  wsum[e] = s;
}

// P = AV @ wo, Qb = Ab @ wo
__global__ void pqb_kernel(const float* __restrict__ AV, const float* __restrict__ Ab,
                           const float* __restrict__ wo, float* __restrict__ P,
                           float* __restrict__ Qb) {
  int idx = blockIdx.x * blockDim.x + threadIdx.x;   // B*D
  if (idx >= 16 * 512) return;
  int e = idx & 511, b = idx >> 9;
  float p0 = 0, p1 = 0, p2 = 0, p3 = 0, qb = 0;
  for (int d = 0; d < 512; ++d) {
    float w = wo[d * 512 + e];
    p0 += AV[(b * 4 + 0) * 512 + d] * w;
    p1 += AV[(b * 4 + 1) * 512 + d] * w;
    p2 += AV[(b * 4 + 2) * 512 + d] * w;
    p3 += AV[(b * 4 + 3) * 512 + d] * w;
    qb += Ab[b * 512 + d] * w;
  }
  P[(b * 4 + 0) * 512 + e] = p0; P[(b * 4 + 1) * 512 + e] = p1;
  P[(b * 4 + 2) * 512 + e] = p2; P[(b * 4 + 3) * 512 + e] = p3;
  Qb[idx] = qb;
}

// x1d (left half of concat buffer, f16)
__global__ void x1d_kernel(const float* __restrict__ fx, const float* __restrict__ P,
                           const float* __restrict__ Qb, const float* __restrict__ bias1d,
                           const float* __restrict__ wsum, const float* __restrict__ bo,
                           half_t* __restrict__ cat) {
  long long idx = (long long)blockIdx.x * blockDim.x + threadIdx.x;  // B*L*D
  if (idx >= (long long)16 * 1024 * 512) return;
  int e = (int)(idx & 511);
  int l = (int)((idx >> 9) & 1023);
  int b = (int)(idx >> 19);
  const float* fb = fx + b * 4096 + l;
  float v = Qb[b * 512 + e] + bias1d[l] * wsum[e] + bo[e];
#pragma unroll
  for (int g = 0; g < 4; ++g) v += fb[g * 1024] * P[(b * 4 + g) * 512 + e];
  cat[((long long)(b * 1024 + l)) * 1024 + e] = (half_t)v;
}

// ---------------- 2D branch: conv3x3 -> tanh offsets -> bilinear ------------
__global__ void off2d_kernel(const float* __restrict__ xn,
                             const float* __restrict__ c1w, const float* __restrict__ c1b,
                             const float* __restrict__ c2w,
                             const float* __restrict__ s_wq2, const float* __restrict__ s_bq2,
                             half_t* __restrict__ xs2h) {
  long long t = (long long)blockIdx.x * blockDim.x + threadIdx.x;
  if (t >= (long long)2032 * 16 * 512) return;
  int j = (int)(t & 511);
  int i = (int)((t >> 9) & 15);
  int patch = (int)(t >> 13);
  int b = patch / 127, p = patch % 127;
  const float* base = xn + ((long long)b * 1024 + p * 8) * 512;  // x4[i][j]
  const float wq2 = *s_wq2, bq2 = *s_bq2;
  float h = c1b[0];
#pragma unroll
  for (int di = 0; di < 3; ++di)
#pragma unroll
    for (int dj = 0; dj < 3; ++dj) {
      int ii = i + di - 1, jj = j + dj - 1;
      if (ii >= 0 && ii < 16 && jj >= 0 && jj < 512)
        h += (wq2 * base[ii * 512 + jj] + bq2) * c1w[di * 3 + dj];
    }
  float dx = tanhf(h * c2w[0]) * 3.0f;
  float dy = tanhf(h * c2w[1]) * 3.0f;
  float gridx = 2.0f * ((float)j + dx) / 511.0f - 1.0f;
  float gridy = 2.0f * ((float)i + dy) / 15.0f - 1.0f;
  float ix = ((gridx + 1.0f) * 512.0f - 1.0f) * 0.5f;
  float iy = ((gridy + 1.0f) * 16.0f - 1.0f) * 0.5f;
  float x0 = floorf(ix), y0 = floorf(iy);
  float wx = ix - x0, wy = iy - y0;
  int xi = (int)x0, yi = (int)y0;
  float acc = 0.0f;
  auto samp = [&](int xx, int yy, float w) {
    if (xx >= 0 && xx < 512 && yy >= 0 && yy < 16) acc += base[yy * 512 + xx] * w;
  };
  samp(xi, yi, (1 - wx) * (1 - wy));
  samp(xi + 1, yi, wx * (1 - wy));
  samp(xi, yi + 1, (1 - wx) * wy);
  samp(xi + 1, yi + 1, wx * wy);
  xs2h[t] = (half_t)acc;
}

// ---------------- per-patch 16x16 attention, one wave per patch -------------
// Writes y TRANSPOSED per batch (yT[b][d][prow], 512 x 2048) so the write GEMM
// consumes it directly as its N-major B operand.
__global__ __launch_bounds__(256) void attn2d_kernel(
    const half_t* __restrict__ xnh, const half_t* __restrict__ xs2h,
    const float* __restrict__ s_wq2, const float* __restrict__ s_bq2,
    const float* __restrict__ s_wk2, const float* __restrict__ s_bk2,
    const float* __restrict__ s_wv2, const float* __restrict__ s_bv2,
    const float* __restrict__ s_wo2, const float* __restrict__ s_bo2,
    const float* __restrict__ bias2d, half_t* __restrict__ yT) {
  __shared__ __align__(16) half_t attnbuf[8][16][16];
  const int lane = threadIdx.x & 31;
  const int wid = threadIdx.x >> 5;
  const int patch = blockIdx.x * 8 + wid;           // 2032 total, exact fit
  const int b = patch / 127, p = patch % 127;
  const float wq2 = *s_wq2, bq2 = *s_bq2, wk2 = *s_wk2, bk2 = *s_bk2;
  const float wv2 = *s_wv2, bv2 = *s_bv2, wo2 = *s_wo2, bo2 = *s_bo2;
  const half_t* xrow = xnh + ((long long)b * 1024 + p * 8) * 512;   // q2 source
  const half_t* srow = xs2h + (long long)patch * (16 * 512);        // sampled
  const int r15 = lane & 15;
  const int alo = (lane < 16) ? 0 : 8;
  const int bk0 = (lane < 16) ? 0 : 8;

  // stage 1: S = q2 . k2^T  (16x16, K=512)
  v8f acc = {};
  for (int kb = 0; kb < 512; kb += 32) {
    union { uint4 u[2]; half_t h[16]; } t0, t1;
    t0.u[0] = *(const uint4*)&xrow[r15 * 512 + kb + alo];
    t0.u[1] = *(const uint4*)&xrow[r15 * 512 + kb + alo + 16];
    t1.u[0] = *(const uint4*)&srow[r15 * 512 + kb + bk0];
    t1.u[1] = *(const uint4*)&srow[r15 * 512 + kb + bk0 + 16];
    v16h a, bf;
#pragma unroll
    for (int s = 0; s < 16; ++s) {
      a[s]  = (half_t)(wq2 * (float)t0.h[s] + bq2);
      bf[s] = (half_t)(wk2 * (float)t1.h[s] + bk2);
    }
    acc = __builtin_amdgcn_wmma_f32_16x16x32_f16(false, a, false, bf, (short)0, acc, false, false);
  }
  // softmax across j (lanes within each 16-lane half)
  float attn[8];
#pragma unroll
  for (int r = 0; r < 8; ++r) {
    float v = acc[r];
    float mx = v;
#pragma unroll
    for (int m = 1; m < 16; m <<= 1) mx = fmaxf(mx, __shfl_xor(mx, m, 32));
    float e = __expf(v - mx);
    float sm = e;
#pragma unroll
    for (int m = 1; m < 16; m <<= 1) sm += __shfl_xor(sm, m, 32);
    attn[r] = e / sm;
  }
  const int crow = (lane < 16) ? 0 : 8;
#pragma unroll
  for (int r = 0; r < 8; ++r) attnbuf[wid][crow + r][r15] = (half_t)attn[r];
  __syncthreads();
  // re-fetch attn in A-layout, zero-padded to K=32
  v16h a2;
#pragma unroll
  for (int s = 0; s < 16; ++s) a2[s] = (lane < 16) ? attnbuf[wid][r15][s] : (half_t)0.0f;
  float b2d[16];
#pragma unroll
  for (int s = 0; s < 16; ++s) b2d[s] = bias2d[s];
  const float msk = (lane < 16) ? 1.0f : 0.0f;      // zero rows k=16..31
  // stage 2: o2 = attn @ v2  (16x512 in 32 WMMA tiles)
  for (int w0 = 0; w0 < 512; w0 += 16) {
    v16h b2;
#pragma unroll
    for (int s = 0; s < 16; ++s) {
      float xv = (float)srow[s * 512 + w0 + r15];
      b2[s] = (half_t)(msk * (wv2 * xv + bv2 + b2d[s]));
    }
    v8f o = {};
    o = __builtin_amdgcn_wmma_f32_16x16x32_f16(false, a2, false, b2, (short)0, o, false, false);
#pragma unroll
    for (int r = 0; r < 8; ++r) {
      int i = crow + r;       // patch row 0..15
      // yT[b][d = w0+r15][prow = p*16+i]
      yT[(((long long)b * 512) + w0 + r15) * 2048 + p * 16 + i] = (half_t)(o[r] * wo2 + bo2);
    }
  }
}

// ---------------- launcher ---------------------------------------------------
extern "C" void kernel_launch(void* const* d_in, const int* in_sizes, int n_in,
                              void* d_out, int out_size, void* d_ws, size_t ws_size,
                              hipStream_t stream) {
  const float* x       = (const float*)d_in[0];
  const float* ln_w    = (const float*)d_in[1];
  const float* ln_b    = (const float*)d_in[2];
  const float* wq      = (const float*)d_in[3];
  const float* bq      = (const float*)d_in[4];
  const float* wk      = (const float*)d_in[5];
  const float* bk      = (const float*)d_in[6];
  const float* wv      = (const float*)d_in[7];
  const float* bv      = (const float*)d_in[8];
  const float* wo      = (const float*)d_in[9];
  const float* bo      = (const float*)d_in[10];
  const float* off1_w  = (const float*)d_in[11];
  const float* off1_b  = (const float*)d_in[12];
  const float* off2_w  = (const float*)d_in[13];
  const float* bias1d  = (const float*)d_in[14];
  const float* wq2     = (const float*)d_in[15];
  const float* bq2     = (const float*)d_in[16];
  const float* wk2     = (const float*)d_in[17];
  const float* bk2     = (const float*)d_in[18];
  const float* wv2     = (const float*)d_in[19];
  const float* bv2     = (const float*)d_in[20];
  const float* wo2     = (const float*)d_in[21];
  const float* bo2     = (const float*)d_in[22];
  const float* offc1_w = (const float*)d_in[23];
  const float* offc1_b = (const float*)d_in[24];
  const float* offc2_w = (const float*)d_in[25];
  const float* bias2d  = (const float*)d_in[26];
  const float* write_w = (const float*)d_in[27];
  const float* write_b = (const float*)d_in[28];
  const float* proj_w  = (const float*)d_in[29];
  const float* proj_b  = (const float*)d_in[30];
  float* out = (float*)d_out;
  (void)in_sizes; (void)n_in; (void)out_size; (void)ws_size;

  char* ws = (char*)d_ws;
  size_t cur = 0;
  auto take = [&](size_t bytes) -> void* {
    void* p = ws + cur;
    cur += (bytes + 255) & ~(size_t)255;
    return p;
  };
  float*  xn   = (float*) take((size_t)16 * 1024 * 512 * 4);
  half_t* xnh  = (half_t*)take((size_t)16 * 1024 * 512 * 2);
  float*  qbuf = (float*) take((size_t)16 * 1024 * 512 * 4);
  half_t* wqT  = (half_t*)take((size_t)512 * 512 * 2);    // wq transposed (NxK)
  half_t* pwT  = (half_t*)take((size_t)512 * 1024 * 2);   // proj_w transposed
  half_t* wwT  = (half_t*)take((size_t)1024 * 2048 * 2);  // write_w^T (A op)
  float*  Wf   = (float*) take((size_t)388 * 4);
  float*  fx   = (float*) take((size_t)16 * 4 * 1024 * 4);
  float*  Kg   = (float*) take((size_t)16 * 4 * 512 * 4);
  float*  Vg   = (float*) take((size_t)16 * 4 * 512 * 4);
  float*  qf   = (float*) take((size_t)16 * 4 * 512 * 4);
  float*  qs   = (float*) take((size_t)16 * 512 * 4);
  float*  AV   = (float*) take((size_t)16 * 4 * 512 * 4);
  float*  Ab   = (float*) take((size_t)16 * 512 * 4);
  float*  P    = (float*) take((size_t)16 * 4 * 512 * 4);
  float*  Qb   = (float*) take((size_t)16 * 512 * 4);
  float*  wsm  = (float*) take((size_t)512 * 4);
  half_t* xs2h = (half_t*)take((size_t)2032 * 16 * 512 * 2);
  half_t* yT   = (half_t*)take((size_t)16 * 512 * 2048 * 2);  // y transposed
  half_t* cat  = (half_t*)take((size_t)16 * 1024 * 1024 * 2);

  // weight prep (every call; deterministic)
  cvtT_f16_kernel<<<(512 * 512 + 255) / 256, 256, 0, stream>>>(wq, wqT, 512, 512);
  cvtT_f16_kernel<<<(512 * 1024 + 255) / 256, 256, 0, stream>>>(proj_w, pwT, 1024, 512);
  wwT_kernel<<<(1024 * 2048 + 255) / 256, 256, 0, stream>>>(write_w, wwT);
  foldw_kernel<<<2, 256, 0, stream>>>(off1_w, off1_b, off2_w, Wf);
  ypad_kernel<<<(16 * 512 * 16 + 255) / 256, 256, 0, stream>>>(yT);

  ln_kernel<<<16 * 1024, 256, 0, stream>>>(x, ln_w, ln_b, xn, xnh);

  // q = xn @ wq + bq   (WMMA + TDM, double-buffered)
  gemm_wmma_kernel<<<dim3(512 / GBN, 16384 / GBM, 1), 256, 0, stream>>>(
      xnh, 512, 0, wqT, 512, 0, (half_t*)nullptr, qbuf, 512, 0,
      bq, nullptr, 16384, 512, 512);

  // collapsed 1D branch
  fx_kernel<<<(16 * 4 * 1024 + 255) / 256, 256, 0, stream>>>(qbuf, Wf, fx);
  kgvg_kernel<<<(16 * 4 * 512 + 255) / 256, 256, 0, stream>>>(xn, wk, wv, Kg, Vg);
  qred_kernel<<<(16 * 512 + 255) / 256, 256, 0, stream>>>(qbuf, fx, qf, qs);
  attn1d_kernel<<<128, 64, 0, stream>>>(qf, qs, Kg, bk, Vg, bv, AV, Ab);
  wsum_kernel<<<2, 256, 0, stream>>>(wo, wsm);
  pqb_kernel<<<(16 * 512 + 255) / 256, 256, 0, stream>>>(AV, Ab, wo, P, Qb);
  x1d_kernel<<<(16 * 1024 * 512 + 255) / 256, 256, 0, stream>>>(fx, P, Qb, bias1d, wsm, bo, cat);

  // 2D branch
  off2d_kernel<<<(2032 * 16 * 512 + 255) / 256, 256, 0, stream>>>(
      xn, offc1_w, offc1_b, offc2_w, wq2, bq2, xs2h);
  attn2d_kernel<<<2032 / 8, 256, 0, stream>>>(
      xnh, xs2h, wq2, bq2, wk2, bk2, wv2, bv2, wo2, bo2, bias2d, yT);

  // x2d = write_w^T @ y + write_b (per batch) -> right half of concat
  gemm_wmma_kernel<<<dim3(512 / GBN, 1024 / GBM, 16), 256, 0, stream>>>(
      wwT, 2048, 0, yT, 2048, (long long)512 * 2048,
      cat + 512, (float*)nullptr, 1024, (long long)1024 * 1024,
      nullptr, write_b, 1024, 512, 2048);

  // out = concat @ proj_w + proj_b
  gemm_wmma_kernel<<<dim3(512 / GBN, 16384 / GBM, 1), 256, 0, stream>>>(
      cat, 1024, 0, pwT, 1024, 0, (half_t*)nullptr, out, 512, 0,
      proj_b, nullptr, 16384, 512, 1024);
}
